// SpatialBlock_18356690223121
// MI455X (gfx1250) — matrix operand order repaired and checked by reading, not compiled, not run on hardware
//
#include <hip/hip_runtime.h>

#define E_DIM   1024
#define N_STATE 16
#define DI      2048
#define DTR     64
#define BATCH   2
#define SEQ     2048
#define NTOK    (BATCH * SEQ)      // 4096 tokens

typedef __attribute__((ext_vector_type(16))) __bf16 v16bf;
typedef __attribute__((ext_vector_type(8)))  __bf16 v8bf;
typedef __attribute__((ext_vector_type(8)))  float  v8f;

// ---------------------------------------------------------------------------
// Materialize TT weight (2 cores, ranks [1,r,1]) as dense W^T in bf16.
// W[(i*m2+j) -> (n*n2+m)] = sum_r g1[i,n,r] * g2[r,j,m]
// Stored transposed: Wt[outIdx][inIdx], outDim = n1*n2 rows, inDim = m1*m2 cols
// so the GEMM B-fragment (per output column) is contiguous in K.
// ---------------------------------------------------------------------------
__global__ void build_tt_w(const float* __restrict__ g1, const float* __restrict__ g2,
                           __bf16* __restrict__ Wt,
                           int m1, int n1, int r, int m2, int n2) {
  int inDim = m1 * m2;
  int total = inDim * n1 * n2;
  int idx = blockIdx.x * blockDim.x + threadIdx.x;
  if (idx >= total) return;
  int col  = idx % inDim;       // input index  i*m2 + j
  int rowO = idx / inDim;       // output index n*n2 + m
  int i = col / m2,  j = col % m2;
  int n = rowO / n2, m = rowO % n2;
  const float* p1 = g1 + (size_t)(i * n1 + n) * r;       // g1[i][n][*]
  float s = 0.f;
  for (int q = 0; q < r; ++q)
    s += p1[q] * g2[((size_t)q * m2 + j) * n2 + m];      // g2[q][j][m]
  Wt[idx] = (__bf16)s;
}

// ---------------------------------------------------------------------------
// Strided f32 -> bf16 (used for x and for the dt_u slice of dtbc)
// ---------------------------------------------------------------------------
__global__ void f32_to_bf16_strided(const float* __restrict__ src, __bf16* __restrict__ dst,
                                    int rows, int cols, int srcStride) {
  int idx = blockIdx.x * blockDim.x + threadIdx.x;
  if (idx >= rows * cols) return;
  int rr = idx / cols, cc = idx % cols;
  dst[idx] = (__bf16)src[(size_t)rr * srcStride + cc];
}

// ---------------------------------------------------------------------------
// WMMA bf16 GEMM:  C[M,N] = A[M,K] * Bt[N,K]^T + bias[N]   (EPI=1: softplus)
// Block = 8 waves; wave tile 32x32 (4 accumulators), block tile 128x64.
// A fragment (16x32 bf16): lane L: row = L&15, K = k + 16*(L>=16) + [0..15]
// B fragment (32x16 bf16): lane L: col = L&15, K = k + 16*(L>=16) + [0..15]
//   -> contiguous 32B load from transposed weights.
// ---------------------------------------------------------------------------
__device__ __forceinline__ float softplus_f(float x) {
  return (x > 20.f) ? x : log1pf(__expf(x));
}

template <int EPI>
__global__ void gemm_wmma_bf16(const __bf16* __restrict__ A,
                               const __bf16* __restrict__ Bt,
                               const float* __restrict__ bias,
                               float* __restrict__ C,
                               int M, int N, int K) {
  const int lane = threadIdx.x & 31;
  const int wave = threadIdx.x >> 5;
  const int waveRow = wave & 3;            // 4 row groups
  const int waveCol = wave >> 2;           // 2 col groups
  const int tileRow = blockIdx.y * 128 + waveRow * 32;
  const int tileCol = blockIdx.x * 64 + waveCol * 32;
  const int l15 = lane & 15;
  const int hi  = lane >> 4;               // 0 or 1

  const int ar0 = tileRow + l15;           // rows for A frag 0
  const int ar1 = tileRow + 16 + l15;      // rows for A frag 1
  const int akOff = hi * 16;
  int bn0 = tileCol + l15;       if (bn0 > N - 1) bn0 = N - 1;   // clamp OOB cols
  int bn1 = tileCol + 16 + l15;  if (bn1 > N - 1) bn1 = N - 1;
  const int bkOff = hi * 16;

  v8f acc[2][2] = {};

  for (int k = 0; k < K; k += 32) {
    const __bf16* pa0 = A + (size_t)ar0 * K + k + akOff;
    const __bf16* pa1 = A + (size_t)ar1 * K + k + akOff;
    v8bf a0lo = *(const v8bf*)(pa0);
    v8bf a0hi = *(const v8bf*)(pa0 + 8);
    v8bf a1lo = *(const v8bf*)(pa1);
    v8bf a1hi = *(const v8bf*)(pa1 + 8);
    v16bf a0 = __builtin_shufflevector(a0lo, a0hi, 0,1,2,3,4,5,6,7,8,9,10,11,12,13,14,15);
    v16bf a1 = __builtin_shufflevector(a1lo, a1hi, 0,1,2,3,4,5,6,7,8,9,10,11,12,13,14,15);

    v16bf b0 = *(const v16bf*)(Bt + (size_t)bn0 * K + k + bkOff);
    v16bf b1 = *(const v16bf*)(Bt + (size_t)bn1 * K + k + bkOff);

    acc[0][0] = __builtin_amdgcn_wmma_f32_16x16x32_bf16(false, a0, false, b0, (short)0, acc[0][0], false, false);
    acc[0][1] = __builtin_amdgcn_wmma_f32_16x16x32_bf16(false, a0, false, b1, (short)0, acc[0][1], false, false);
    acc[1][0] = __builtin_amdgcn_wmma_f32_16x16x32_bf16(false, a1, false, b0, (short)0, acc[1][0], false, false);
    acc[1][1] = __builtin_amdgcn_wmma_f32_16x16x32_bf16(false, a1, false, b1, (short)0, acc[1][1], false, false);
  }

  // Epilogue. C/D layout: VGPR v, lanes 0-15 -> M=v, lanes 16-31 -> M=v+8; N=lane&15.
  #pragma unroll
  for (int i = 0; i < 2; ++i) {
    #pragma unroll
    for (int j = 0; j < 2; ++j) {
      int col = tileCol + j * 16 + l15;
      if (col < N) {
        float bv = bias[col];
        int rbase = tileRow + i * 16 + hi * 8;
        #pragma unroll
        for (int v = 0; v < 8; ++v) {
          float val = acc[i][j][v] + bv;
          if (EPI == 1) val = softplus_f(val);
          C[(size_t)(rbase + v) * N + col] = val;
        }
      }
    }
  }
}

// ---------------------------------------------------------------------------
// Depthwise conv1d (K=4, pad left 1 / right 2) + bias + SiLU, gated by z.
// xz layout: (NTOK, 2*DI): first DI = x_conv, last DI = z.
// Writes u (f32, scan input) and u16 (bf16, GEMM input).
// ---------------------------------------------------------------------------
__global__ void conv_silu_gate(const float* __restrict__ xz,
                               const float* __restrict__ cw,  // (4,1,DI)
                               const float* __restrict__ cb,  // (DI)
                               float* __restrict__ u, __bf16* __restrict__ u16) {
  int idx = blockIdx.x * blockDim.x + threadIdx.x;     // over BATCH*SEQ*DI
  if (idx >= NTOK * DI) return;
  int c = idx % DI;
  int t = idx / DI;
  int l = t % SEQ;
  float acc = cb[c];
  #pragma unroll
  for (int k = 0; k < 4; ++k) {
    int ll = l + k - 1;
    if (ll >= 0 && ll < SEQ)
      acc += xz[((size_t)(t + (ll - l))) * (2 * DI) + c] * cw[k * DI + c];
  }
  float z = xz[(size_t)t * (2 * DI) + DI + c];
  float s = acc / (1.f + __expf(-acc));                // SiLU
  float uu = s * z;
  u[idx]   = uu;
  u16[idx] = (__bf16)uu;
}

// ---------------------------------------------------------------------------
// Selective scan. One lane per (b, d, n); 16 lanes per channel group.
// h_{l} = exp(dt*A)*h_{l-1} + dt*u*B[n];  y = sum_n h*C[n] + u*D  (bf16 out).
// xor-shuffle reduction (masks 1,2,4,8) stays within each 16-lane half.
// ---------------------------------------------------------------------------
__global__ void selective_scan_kernel(const float* __restrict__ u,
                                      const float* __restrict__ dt,
                                      const float* __restrict__ dtbc,   // (NTOK, 96)
                                      const float* __restrict__ A_log,  // (DI, 16)
                                      const float* __restrict__ Dp,     // (DI)
                                      __bf16* __restrict__ ssm16) {
  int tid = blockIdx.x * blockDim.x + threadIdx.x;     // BATCH*DI*16 = 65536
  int n = tid & 15;
  int d = (tid >> 4) & (DI - 1);
  int b = tid >> 15;
  float Aa = -__expf(A_log[d * N_STATE + n]);
  float Dd = Dp[d];
  float h = 0.f;
  const float* dtp  = dt   + (size_t)b * SEQ * DI + d;
  const float* up   = u    + (size_t)b * SEQ * DI + d;
  const float* bc   = dtbc + (size_t)b * SEQ * 96;
  __bf16* outp      = ssm16 + (size_t)b * SEQ * DI + d;

  for (int l = 0; l < SEQ; ++l) {
    float dtv = dtp[(size_t)l * DI];
    float uv  = up[(size_t)l * DI];
    float Bv  = bc[l * 96 + DTR + n];
    float Cv  = bc[l * 96 + DTR + N_STATE + n];
    float dA  = __expf(dtv * Aa);
    h = dA * h + dtv * uv * Bv;
    float p = h * Cv;
    p += __shfl_xor(p, 1, 32);
    p += __shfl_xor(p, 2, 32);
    p += __shfl_xor(p, 4, 32);
    p += __shfl_xor(p, 8, 32);
    if (n == 0) outp[(size_t)l * DI] = (__bf16)(p + uv * Dd);
  }
}

// ---------------------------------------------------------------------------
// Host orchestration
// ---------------------------------------------------------------------------
static inline size_t align256(size_t x) { return (x + 255) & ~(size_t)255; }

extern "C" void kernel_launch(void* const* d_in, const int* in_sizes, int n_in,
                              void* d_out, int out_size, void* d_ws, size_t ws_size,
                              hipStream_t stream) {
  const float* x      = (const float*)d_in[0];
  const float* in_g1  = (const float*)d_in[1];
  const float* in_g2  = (const float*)d_in[2];
  const float* in_b   = (const float*)d_in[3];
  const float* conv_w = (const float*)d_in[4];
  const float* conv_b = (const float*)d_in[5];
  const float* xp_g1  = (const float*)d_in[6];
  const float* xp_g2  = (const float*)d_in[7];
  const float* xp_b   = (const float*)d_in[8];
  const float* dt_g1  = (const float*)d_in[9];
  const float* dt_g2  = (const float*)d_in[10];
  const float* dt_b   = (const float*)d_in[11];
  const float* A_log  = (const float*)d_in[12];
  const float* D_par  = (const float*)d_in[13];
  const float* out_g1 = (const float*)d_in[14];
  const float* out_g2 = (const float*)d_in[15];
  const float* out_b  = (const float*)d_in[16];
  float* out = (float*)d_out;

  // Workspace layout
  char* ws = (char*)d_ws;
  size_t off = 0;
  auto carve = [&](size_t bytes) { char* p = ws + off; off += align256(bytes); return p; };

  __bf16* W_in  = (__bf16*)carve((size_t)4096 * 1024 * 2);  // (n1*n2=4096) x (K=1024)
  __bf16* W_xp  = (__bf16*)carve((size_t)96   * 2048 * 2);
  __bf16* W_dt  = (__bf16*)carve((size_t)2048 * 64   * 2);
  __bf16* W_out = (__bf16*)carve((size_t)1024 * 2048 * 2);
  __bf16* x16   = (__bf16*)carve((size_t)NTOK * E_DIM * 2);
  float*  xz    = (float*) carve((size_t)NTOK * 2 * DI * 4);
  float*  u     = (float*) carve((size_t)NTOK * DI * 4);
  __bf16* u16   = (__bf16*)carve((size_t)NTOK * DI * 2);
  float*  dtbc  = (float*) carve((size_t)NTOK * 96 * 4);
  __bf16* dtu16 = (__bf16*)carve((size_t)NTOK * DTR * 2);
  float*  dtf   = (float*) carve((size_t)NTOK * DI * 4);
  __bf16* ssm16 = (__bf16*)carve((size_t)NTOK * DI * 2);
  (void)ws_size;

  const int T = 256;
  auto blocks = [&](long total) { return (unsigned)((total + T - 1) / T); };

  // 1. Materialize TT weights (transposed, bf16)
  build_tt_w<<<blocks((long)4096 * 1024), T, 0, stream>>>(in_g1,  in_g2,  W_in,  32, 64, 16, 32, 64);
  build_tt_w<<<blocks((long)96   * 2048), T, 0, stream>>>(xp_g1,  xp_g2,  W_xp,  32,  8, 16, 64, 12);
  build_tt_w<<<blocks((long)2048 * 64  ), T, 0, stream>>>(dt_g1,  dt_g2,  W_dt,   8, 32, 16,  8, 64);
  build_tt_w<<<blocks((long)1024 * 2048), T, 0, stream>>>(out_g1, out_g2, W_out, 32, 32, 16, 64, 32);

  // 2. x -> bf16
  f32_to_bf16_strided<<<blocks((long)NTOK * E_DIM), T, 0, stream>>>(x, x16, NTOK, E_DIM, E_DIM);

  // 3. in_proj GEMM: xz = x16 * W_in^T + in_b   (4096 x 4096, K=1024)
  gemm_wmma_bf16<0><<<dim3(4096 / 64, NTOK / 128), T, 0, stream>>>(x16, W_in, in_b, xz, NTOK, 4096, 1024);

  // 4. depthwise conv + SiLU gate -> u (f32) and u16 (bf16)
  conv_silu_gate<<<blocks((long)NTOK * DI), T, 0, stream>>>(xz, conv_w, conv_b, u, u16);

  // 5. x_proj GEMM: dtbc = u16 * W_xp^T + xp_b  (4096 x 96, K=2048)
  gemm_wmma_bf16<0><<<dim3((96 + 63) / 64, NTOK / 128), T, 0, stream>>>(u16, W_xp, xp_b, dtbc, NTOK, 96, 2048);

  // 6. dt_u slice -> bf16 (first 64 cols of dtbc)
  f32_to_bf16_strided<<<blocks((long)NTOK * DTR), T, 0, stream>>>(dtbc, dtu16, NTOK, DTR, 96);

  // 7. dt_proj GEMM + softplus: dt = softplus(dtu16 * W_dt^T + dt_b) (4096 x 2048, K=64)
  gemm_wmma_bf16<1><<<dim3(2048 / 64, NTOK / 128), T, 0, stream>>>(dtu16, W_dt, dt_b, dtf, NTOK, 2048, 64);

  // 8. selective scan -> ssm16 (bf16)
  selective_scan_kernel<<<blocks((long)BATCH * DI * 16), T, 0, stream>>>(u, dtf, dtbc, A_log, D_par, ssm16);

  // 9. out_proj GEMM: out = ssm16 * W_out^T + out_b (4096 x 1024, K=2048)
  gemm_wmma_bf16<0><<<dim3(1024 / 64, NTOK / 128), T, 0, stream>>>(ssm16, W_out, out_b, out, NTOK, 1024, 2048);
}
